// pointnet2_perceptual_backbone_59030030516577
// MI455X (gfx1250) — compile-verified, matrix operand order checked by reading, not run on hardware
//
#include <hip/hip_runtime.h>
#include <hip/hip_bf16.h>

// ---------------------------------------------------------------------------
// PointNet++ MSG backbone for MI455X (gfx1250, wave32, WMMA).
// GEMM core: v_wmma_f32_16x16x32_f16, one wave per 16x(16*NT) output strip —
// the A fragment is loaded once per k-chunk and reused across NT accumulators.
// ---------------------------------------------------------------------------

typedef _Float16 h8   __attribute__((ext_vector_type(8)));
typedef _Float16 v16h __attribute__((ext_vector_type(16)));
typedef float    v8f  __attribute__((ext_vector_type(8)));

#define BB   4
#define NN   8192
#define S1   2048
#define S2   512

static inline int pad32(int x) { return (x + 31) & ~31; }

// ---------------------------------- utility -------------------------------
__global__ void k_zero_f32(float* p, long long n) {
  long long i  = (long long)blockIdx.x * blockDim.x + threadIdx.x;
  long long st = (long long)gridDim.x * blockDim.x;
  for (; i < n; i += st) p[i] = 0.f;
}

// voxel indices -> xyz  (coords reversed: (c2,c1,c0), *0.1 + range_min)
__global__ void k_compute_xyz(const int* __restrict__ ind, float* __restrict__ xyz, int total) {
  int i = blockIdx.x * blockDim.x + threadIdx.x;
  if (i >= total) return;
  float c0 = (float)ind[i * 4 + 1];
  float c1 = (float)ind[i * 4 + 2];
  float c2 = (float)ind[i * 4 + 3];
  xyz[i * 3 + 0] = (c2 + 0.5f) * 0.1f + 0.0f;
  xyz[i * 3 + 1] = (c1 + 0.5f) * 0.1f - 40.0f;
  xyz[i * 3 + 2] = (c0 + 0.5f) * 0.1f - 3.0f;
}

// ------------------------------ farthest point sampling --------------------
// one block per batch; dist[] lives in LDS; block argmax, ties -> lowest idx.
__global__ void k_fps(const float* __restrict__ xyz, float* __restrict__ newxyz,
                      int N, int npoint) {
  extern __shared__ float smem[];
  float* dist = smem;                 // N
  float* rval = smem + N;             // blockDim
  int*   ridx = (int*)(rval + blockDim.x);
  __shared__ float cen[3];
  __shared__ int   farS;
  const int b = blockIdx.x;
  const float* X = xyz + (size_t)b * N * 3;
  for (int i = threadIdx.x; i < N; i += blockDim.x) dist[i] = 1e10f;
  if (threadIdx.x == 0) farS = 0;
  __syncthreads();
  for (int it = 0; it < npoint; ++it) {
    if (threadIdx.x == 0) {
      int f = farS;
      float cx = X[f * 3], cy = X[f * 3 + 1], cz = X[f * 3 + 2];
      cen[0] = cx; cen[1] = cy; cen[2] = cz;
      float* o = newxyz + ((size_t)b * npoint + it) * 3;
      o[0] = cx; o[1] = cy; o[2] = cz;
    }
    __syncthreads();
    float cx = cen[0], cy = cen[1], cz = cen[2];
    float bv = -1.f; int bi = 0;
    for (int i = threadIdx.x; i < N; i += blockDim.x) {
      float dx = X[i * 3] - cx, dy = X[i * 3 + 1] - cy, dz = X[i * 3 + 2] - cz;
      float d  = dx * dx + dy * dy + dz * dz;
      float nd = fminf(dist[i], d);
      dist[i]  = nd;
      if (nd > bv) { bv = nd; bi = i; }
    }
    rval[threadIdx.x] = bv; ridx[threadIdx.x] = bi;
    __syncthreads();
    for (int off = blockDim.x >> 1; off > 0; off >>= 1) {
      if (threadIdx.x < off) {
        float v2 = rval[threadIdx.x + off]; int i2 = ridx[threadIdx.x + off];
        float v1 = rval[threadIdx.x];       int i1 = ridx[threadIdx.x];
        if (v2 > v1 || (v2 == v1 && i2 < i1)) { rval[threadIdx.x] = v2; ridx[threadIdx.x] = i2; }
      }
      __syncthreads();
    }
    if (threadIdx.x == 0) farS = ridx[0];
    __syncthreads();
  }
}

// ------------------------------- ball query --------------------------------
// one wave32 per query: keep first K in-range indices in ascending order
// (== sort-then-truncate in the reference), pad with first hit.
__global__ void k_ball_query(const float* __restrict__ xyz, const float* __restrict__ newxyz,
                             int* __restrict__ gi, int Bq, int S, int N, int K, float r2) {
  int gw   = (blockIdx.x * blockDim.x + threadIdx.x) >> 5;
  int lane = threadIdx.x & 31;
  if (gw >= Bq) return;
  int b = gw / S;
  const float* X = xyz + (size_t)b * N * 3;
  const float* q = newxyz + (size_t)gw * 3;
  float qx = q[0], qy = q[1], qz = q[2];
  int* out = gi + (size_t)gw * K;
  int  cnt = 0, first = 0;
  bool haveFirst = false;
  for (int base = 0; base < N; base += 32) {
    if (cnt >= K) break;
    int  i  = base + lane;
    bool in = false;
    if (i < N) {
      float dx = X[i * 3] - qx, dy = X[i * 3 + 1] - qy, dz = X[i * 3 + 2] - qz;
      in = (dx * dx + dy * dy + dz * dz) <= r2;
    }
    unsigned mask = __builtin_amdgcn_ballot_w32(in);
    if (!haveFirst && mask) { first = base + (__ffs(mask) - 1); haveFirst = true; }
    if (in) {
      int pos = cnt + __popc(mask & ((1u << lane) - 1u));
      if (pos < K) out[pos] = i;
    }
    cnt += __popc(mask);
  }
  if (cnt > K) cnt = K;
  for (int p = cnt + lane; p < K; p += 32) out[p] = first;
}

// ------------------------------- grouping ----------------------------------
__global__ void k_gather_groups(const float* __restrict__ xyz, const float* __restrict__ newxyz,
                                const float* __restrict__ pts, const int* __restrict__ gi,
                                _Float16* __restrict__ A, int BS, int S, int N, int K,
                                int Cin, int Kp) {
  int row = blockIdx.x * blockDim.x + threadIdx.x;
  if (row >= BS * K) return;
  int k = row % K;
  int g = row / K;
  int b = g / S;
  int gidx = gi[(size_t)g * K + k];
  const float* p = pts + ((size_t)b * N + gidx) * Cin;
  _Float16* dst = A + (size_t)row * Kp;
  for (int c = 0; c < Cin; ++c) dst[c] = (_Float16)p[c];
  const float* px = xyz + ((size_t)b * N + gidx) * 3;
  const float* nx = newxyz + (size_t)g * 3;
  dst[Cin + 0] = (_Float16)(px[0] - nx[0]);
  dst[Cin + 1] = (_Float16)(px[1] - nx[1]);
  dst[Cin + 2] = (_Float16)(px[2] - nx[2]);
  for (int c = Cin + 3; c < Kp; ++c) dst[c] = (_Float16)0.f;
}

// SA3: g = concat([xyz, points]) per point (xyz FIRST here, per reference)
__global__ void k_gather_all(const float* __restrict__ xyz, const float* __restrict__ pts,
                             _Float16* __restrict__ A, int rows, int Cin, int Kp) {
  int r = blockIdx.x * blockDim.x + threadIdx.x;
  if (r >= rows) return;
  _Float16* dst = A + (size_t)r * Kp;
  dst[0] = (_Float16)xyz[r * 3 + 0];
  dst[1] = (_Float16)xyz[r * 3 + 1];
  dst[2] = (_Float16)xyz[r * 3 + 2];
  const float* p = pts + (size_t)r * Cin;
  for (int c = 0; c < Cin; ++c) dst[3 + c] = (_Float16)p[c];
  for (int c = 3 + Cin; c < Kp; ++c) dst[c] = (_Float16)0.f;
}

// weights (cout x cin, f32) -> f16, zero-padded to Kp columns
__global__ void k_pack_w(const float* __restrict__ w, _Float16* __restrict__ wh,
                         int cout, int cin, int Kp) {
  int i = blockIdx.x * blockDim.x + threadIdx.x;
  if (i >= cout * Kp) return;
  int c = i / Kp, k = i % Kp;
  wh[i] = (k < cin) ? (_Float16)w[(size_t)c * cin + k] : (_Float16)0.f;
}

// ------------------------------- WMMA GEMM ---------------------------------
// H[m][n] = sum_k A[m][k]*W[n][k] + bias[n].
// One wave per 16x(16*NT) strip: A fragment loaded once per 32-wide k-chunk,
// reused by NT WMMAs into NT resident f32 accumulators (NT*8 VGPRs).
// Operand layout per CDNA5 ISA 16-bit A/B: lane L -> row/col L%16,
// lanes 0-15 hold k {0..7,16..23}, lanes 16-31 hold k {8..15,24..31}.
template <int NT>
__global__ void k_wmma_gemm(const _Float16* __restrict__ A, const _Float16* __restrict__ W,
                            const float* __restrict__ bias, float* __restrict__ H,
                            int Mtiles, int NtGroups, int Kp, int cout) {
  int wave = (int)((blockIdx.x * blockDim.x + threadIdx.x) >> 5);
  int lane = threadIdx.x & 31;
  if (wave >= Mtiles * NtGroups) return;
  int mt  = wave / NtGroups;
  int ng  = wave - mt * NtGroups;
  int nt0 = ng * NT;
  int l16  = lane & 15;
  int koff = (lane < 16) ? 0 : 8;
  const _Float16* arow = A + (size_t)(mt * 16 + l16) * Kp + koff;
  const _Float16* brow[NT];
#pragma unroll
  for (int j = 0; j < NT; ++j)
    brow[j] = W + (size_t)((nt0 + j) * 16 + l16) * Kp + koff;
  v8f c[NT];
#pragma unroll
  for (int j = 0; j < NT; ++j) c[j] = (v8f){};
  union U { v16h v; struct { h8 lo; h8 hi; } p; };
  for (int kk = 0; kk < Kp; kk += 32) {
    __builtin_prefetch(arow + kk + 64, 0, 1);
    U a;
    a.p.lo = *(const h8*)(arow + kk);
    a.p.hi = *(const h8*)(arow + kk + 16);
#pragma unroll
    for (int j = 0; j < NT; ++j) {
      U b;
      b.p.lo = *(const h8*)(brow[j] + kk);
      b.p.hi = *(const h8*)(brow[j] + kk + 16);
      c[j] = __builtin_amdgcn_wmma_f32_16x16x32_f16(false, a.v, false, b.v,
                                                    (short)0, c[j], false, false);
    }
  }
  int row0 = mt * 16 + ((lane < 16) ? 0 : 8);
#pragma unroll
  for (int j = 0; j < NT; ++j) {
    int col = (nt0 + j) * 16 + l16;
    float bv = bias[col];
    float* Hp = H + (size_t)row0 * cout + col;
#pragma unroll
    for (int g2 = 0; g2 < 8; ++g2) Hp[(size_t)g2 * cout] = c[j][g2] + bv;
  }
}

// ------------------------- BN stats + epilogues -----------------------------
#define STATS_ROWS 128
__global__ void k_stats(const float* __restrict__ H, float* __restrict__ acc, int M, int cout) {
  __shared__ float ssum[1024];
  __shared__ float ssq[1024];
  for (int c = threadIdx.x; c < cout; c += blockDim.x) { ssum[c] = 0.f; ssq[c] = 0.f; }
  __syncthreads();
  long long r0    = (long long)blockIdx.x * STATS_ROWS;
  long long elems = (long long)STATS_ROWS * cout;
  long long lim   = ((long long)M - r0) * cout;
  if (lim > elems) lim = elems;
  const float* base = H + r0 * cout;
  for (long long i = threadIdx.x; i < lim; i += blockDim.x) {
    float v = base[i];
    int   c = (int)(i % cout);
    atomicAdd(&ssum[c], v);
    atomicAdd(&ssq[c], v * v);
  }
  __syncthreads();
  for (int c = threadIdx.x; c < cout; c += blockDim.x) {
    atomicAdd(&acc[c], ssum[c]);
    atomicAdd(&acc[cout + c], ssq[c]);
  }
}

// mid layer: BN+ReLU -> f16 operand for next GEMM (zero pad cols)
__global__ void k_bn_relu_f16(const float* __restrict__ H, const float* __restrict__ acc,
                              const float* __restrict__ gamma, const float* __restrict__ beta,
                              _Float16* __restrict__ Anext, long long M, int cout,
                              int KpNext, float invM) {
  long long i  = (long long)blockIdx.x * blockDim.x + threadIdx.x;
  long long n  = M * KpNext;
  long long st = (long long)gridDim.x * blockDim.x;
  for (; i < n; i += st) {
    int c = (int)(i % KpNext);
    if (c >= cout) { Anext[i] = (_Float16)0.f; continue; }
    long long r = i / KpNext;
    float h    = H[r * cout + c];
    float mean = acc[c] * invM;
    float var  = acc[cout + c] * invM - mean * mean;
    float y    = (h - mean) * rsqrtf(var + 1e-5f) * gamma[c] + beta[c];
    Anext[i]   = (_Float16)(y > 0.f ? y : 0.f);
  }
}

// last layer: BN+ReLU, max over K, write row-major feats (ws) + transposed d_out
__global__ void k_bn_relu_maxpool(const float* __restrict__ H, const float* __restrict__ acc,
                                  const float* __restrict__ gamma, const float* __restrict__ beta,
                                  float* __restrict__ featws, float* __restrict__ dout,
                                  int BS, int K, int cout, int Ctot, int coff, int S,
                                  long long outoff, float invM) {
  int idx = blockIdx.x * blockDim.x + threadIdx.x;
  if (idx >= BS * cout) return;
  int g = idx / cout, c = idx % cout;
  float mean = acc[c] * invM;
  float var  = acc[cout + c] * invM - mean * mean;
  float sc   = rsqrtf(var + 1e-5f) * gamma[c];
  float sh   = beta[c] - mean * sc;
  float m    = -3.4e38f;
  const float* base = H + (size_t)g * K * cout + c;
  for (int k = 0; k < K; ++k) {
    float y = base[(size_t)k * cout] * sc + sh;
    y = y > 0.f ? y : 0.f;
    if (y > m) m = y;
  }
  if (featws) featws[(size_t)g * Ctot + coff + c] = m;
  int b = g / S, s = g - b * S;
  dout[outoff + ((size_t)b * Ctot + coff + c) * S + s] = m;
}

// =========================== host orchestration =============================
struct LP { const float* w; const float* b; const float* g; const float* be; int cin; int cout; };

static void get_layer(void* const* d_in, const int* in_sizes, int base, int cin, int cout, LP& L) {
  // robust to either pytree-sorted (b,beta,gamma,w) or insertion (w,b,gamma,beta)
  int wi = -1;
  for (int j = 0; j < 4; ++j)
    if (in_sizes[base + j] == cin * cout) { wi = j; break; }
  if (wi == 0) {
    L.w  = (const float*)d_in[base + 0];
    L.b  = (const float*)d_in[base + 1];
    L.g  = (const float*)d_in[base + 2];
    L.be = (const float*)d_in[base + 3];
  } else {
    L.b  = (const float*)d_in[base + 0];
    L.be = (const float*)d_in[base + 1];
    L.g  = (const float*)d_in[base + 2];
    L.w  = (const float*)d_in[base + 3];
  }
  L.cin = cin; L.cout = cout;
}

static void launch_gemm(hipStream_t st, const _Float16* A, const _Float16* W, const float* bias,
                        float* H, int Mt, int Ntiles, int Kp, int cout) {
  int NT = (Ntiles % 4 == 0) ? 4 : (Ntiles % 3 == 0) ? 3 : (Ntiles % 2 == 0) ? 2 : 1;
  int ng = Ntiles / NT;
  long long waves  = (long long)Mt * ng;
  int       blocks = (int)((waves + 7) / 8);
  switch (NT) {
    case 4: k_wmma_gemm<4><<<blocks, 256, 0, st>>>(A, W, bias, H, Mt, ng, Kp, cout); break;
    case 3: k_wmma_gemm<3><<<blocks, 256, 0, st>>>(A, W, bias, H, Mt, ng, Kp, cout); break;
    case 2: k_wmma_gemm<2><<<blocks, 256, 0, st>>>(A, W, bias, H, Mt, ng, Kp, cout); break;
    default: k_wmma_gemm<1><<<blocks, 256, 0, st>>>(A, W, bias, H, Mt, ng, Kp, cout); break;
  }
}

static void conv_layer(hipStream_t st, const LP& L, const _Float16* Ain, int M, int Kp,
                       _Float16* Wh, float* H, float* acc) {
  { int n = L.cout * Kp; k_pack_w<<<(n + 255) / 256, 256, 0, st>>>(L.w, Wh, L.cout, L.cin, Kp); }
  launch_gemm(st, Ain, Wh, L.b, H, M / 16, L.cout / 16, Kp, L.cout);
  k_zero_f32<<<(2 * L.cout + 255) / 256, 256, 0, st>>>(acc, 2 * L.cout);
  int sb = (M + STATS_ROWS - 1) / STATS_ROWS;
  k_stats<<<sb, 256, 0, st>>>(H, acc, M, L.cout);
}

static void run_branch(hipStream_t st, const float* srcxyz, const float* newxyz, const float* pts,
                       const int* gi, int S, int N, int K, int Cin, const LP* L,
                       _Float16* A0, _Float16* A1, _Float16* Wh, float* H, float* acc,
                       float* featws, float* dout, int Ctot, int coff, long long outoff) {
  int M  = BB * S * K;
  int Kp = pad32(Cin + 3);
  k_gather_groups<<<(M + 255) / 256, 256, 0, st>>>(srcxyz, newxyz, pts, gi, A0, BB * S, S, N, K, Cin, Kp);
  const _Float16* Ain  = A0;
  _Float16*       Aout = A1;
  for (int li = 0; li < 3; ++li) {
    conv_layer(st, L[li], Ain, M, Kp, Wh, H, acc);
    float invM = 1.f / (float)M;
    if (li < 2) {
      int KpN     = pad32(L[li].cout);
      long long n = (long long)M * KpN;
      k_bn_relu_f16<<<(int)((n + 255) / 256), 256, 0, st>>>(H, acc, L[li].g, L[li].be, Aout,
                                                            M, L[li].cout, KpN, invM);
      const _Float16* t = Aout; Aout = (Aout == A1) ? A0 : A1; Ain = t;
      Kp = KpN;
    } else {
      int n = BB * S * L[li].cout;
      k_bn_relu_maxpool<<<(n + 255) / 256, 256, 0, st>>>(H, acc, L[li].g, L[li].be, featws, dout,
                                                         BB * S, K, L[li].cout, Ctot, coff, S,
                                                         outoff, invM);
    }
  }
}

extern "C" void kernel_launch(void* const* d_in, const int* in_sizes, int n_in,
                              void* d_out, int out_size, void* d_ws, size_t ws_size,
                              hipStream_t stream) {
  const float* feat1 = (const float*)d_in[0];
  const float* feat2 = (const float*)d_in[1];
  const int*   ind   = (const int*)d_in[2];
  float*       dout  = (float*)d_out;

  // ----- parameters (sa1: 3 branches x 3 layers, sa2: same, sa3: 3 layers)
  static const int sa1ch[3][3] = {{32, 32, 64}, {64, 64, 128}, {64, 96, 128}};
  static const int sa2ch[3][3] = {{64, 64, 128}, {128, 128, 256}, {128, 128, 256}};
  static const int sa3ch[3]    = {256, 512, 1024};
  LP sa1[3][3], sa2[3][3], sa3[3];
  for (int bi = 0; bi < 3; ++bi) {
    int cin = 6;
    for (int li = 0; li < 3; ++li) { get_layer(d_in, in_sizes, 4 + (bi * 3 + li) * 4, cin, sa1ch[bi][li], sa1[bi][li]); cin = sa1ch[bi][li]; }
  }
  for (int bi = 0; bi < 3; ++bi) {
    int cin = 323;
    for (int li = 0; li < 3; ++li) { get_layer(d_in, in_sizes, 40 + (bi * 3 + li) * 4, cin, sa2ch[bi][li], sa2[bi][li]); cin = sa2ch[bi][li]; }
  }
  { int cin = 643;
    for (int li = 0; li < 3; ++li) { get_layer(d_in, in_sizes, 76 + li * 4, cin, sa3ch[li], sa3[li]); cin = sa3ch[li]; } }

  // ----- workspace carve
  char*  ws  = (char*)d_ws;
  size_t off = 0;
  auto carve = [&](size_t bytes) -> void* { void* p = ws + off; off = (off + bytes + 255) & ~(size_t)255; return p; };
  float* xyz = (float*)carve((size_t)BB * NN * 3 * 4);
  float* nx1 = (float*)carve((size_t)BB * S1 * 3 * 4);
  float* nx2 = (float*)carve((size_t)BB * S2 * 3 * 4);
  static const int   K1[3] = {16, 32, 128};
  static const int   K2[3] = {32, 64, 128};
  static const float R1[3] = {0.4f, 0.8f, 1.2f};
  static const float R2[3] = {0.8f, 1.2f, 2.4f};
  int* gi1[3]; for (int i = 0; i < 3; ++i) gi1[i] = (int*)carve((size_t)BB * S1 * K1[i] * 4);
  int* gi2[3]; for (int i = 0; i < 3; ++i) gi2[i] = (int*)carve((size_t)BB * S2 * K2[i] * 4);
  float* l1p[2]; for (int p = 0; p < 2; ++p) l1p[p] = (float*)carve((size_t)BB * S1 * 320 * 4);
  float* l2p[2]; for (int p = 0; p < 2; ++p) l2p[p] = (float*)carve((size_t)BB * S2 * 640 * 4);
  float*    acc = (float*)carve(2 * 1024 * 4);
  _Float16* Wh  = (_Float16*)carve((size_t)1024 * 672 * 2);
  const size_t ASZ = (size_t)1048576 * 96 * 2;   // max f16 activation buffer
  const size_t HSZ = (size_t)1048576 * 128 * 4;  // max f32 GEMM output
  _Float16* Aa = (_Float16*)carve(ASZ);
  _Float16* Ab = (_Float16*)carve(ASZ);
  float*    H  = (float*)carve(HSZ);

  // output offsets (floats, tuple concat order)
  const long long oL1[2] = {0LL, 3936256LL};
  const long long oL2[2] = {2621440LL, 6557696LL};
  const long long oL3[2] = {3932160LL, 7868416LL};

  // ----- geometry (shared by both feature passes)
  k_compute_xyz<<<(BB * NN + 255) / 256, 256, 0, stream>>>(ind, xyz, BB * NN);
  { size_t shm = (size_t)(NN + 256) * 4 + 256 * 4;
    k_fps<<<BB, 256, shm, stream>>>(xyz, nx1, NN, S1); }
  { size_t shm = (size_t)(S1 + 256) * 4 + 256 * 4;
    k_fps<<<BB, 256, shm, stream>>>(nx1, nx2, S1, S2); }
  for (int i = 0; i < 3; ++i) {
    int thr = BB * S1 * 32;
    k_ball_query<<<(thr + 255) / 256, 256, 0, stream>>>(xyz, nx1, gi1[i], BB * S1, S1, NN, K1[i], R1[i] * R1[i]);
  }
  for (int i = 0; i < 3; ++i) {
    int thr = BB * S2 * 32;
    k_ball_query<<<(thr + 255) / 256, 256, 0, stream>>>(nx1, nx2, gi2[i], BB * S2, S2, S1, K2[i], R2[i] * R2[i]);
  }

  // ----- two feature passes (shared centroids / ball-query indices)
  static const int off1[3] = {0, 64, 192};
  static const int off2[3] = {0, 128, 384};
  for (int p = 0; p < 2; ++p) {
    const float* feats = (p == 0) ? feat1 : feat2;
    // SA1 (MSG): points = raw features (3 ch)
    for (int bi = 0; bi < 3; ++bi)
      run_branch(stream, xyz, nx1, feats, gi1[bi], S1, NN, K1[bi], 3, sa1[bi],
                 Aa, Ab, Wh, H, acc, l1p[p], dout, 320, off1[bi], oL1[p]);
    // SA2 (MSG): points = l1p (320 ch)
    for (int bi = 0; bi < 3; ++bi)
      run_branch(stream, nx1, nx2, l1p[p], gi2[bi], S2, S1, K2[bi], 320, sa2[bi],
                 Aa, Ab, Wh, H, acc, l2p[p], dout, 640, off2[bi], oL2[p]);
    // SA3 (group-all): rows = B*S2, cin = 3 + 640 -> pad 672
    {
      int M = BB * S2, Kp = pad32(643);
      k_gather_all<<<(M + 255) / 256, 256, 0, stream>>>(nx2, l2p[p], Aa, M, 640, Kp);
      const _Float16* Ain  = Aa;
      _Float16*       Aout = Ab;
      for (int li = 0; li < 3; ++li) {
        conv_layer(stream, sa3[li], Ain, M, Kp, Wh, H, acc);
        float invM = 1.f / (float)M;
        if (li < 2) {
          int KpN     = pad32(sa3[li].cout);
          long long n = (long long)M * KpN;
          k_bn_relu_f16<<<(int)((n + 255) / 256), 256, 0, stream>>>(H, acc, sa3[li].g, sa3[li].be,
                                                                    Aout, M, sa3[li].cout, KpN, invM);
          const _Float16* t = Aout; Aout = (Aout == Ab) ? Aa : Ab; Ain = t;
          Kp = KpN;
        } else {
          int n = BB * sa3[li].cout;  // BS = B, K = S2, S = 1
          k_bn_relu_maxpool<<<(n + 255) / 256, 256, 0, stream>>>(H, acc, sa3[li].g, sa3[li].be,
                                                                 nullptr, dout, BB, S2, sa3[li].cout,
                                                                 1024, 0, 1, oL3[p], invM);
        }
      }
    }
  }
}
